// NeuralODE_73959336837611
// MI455X (gfx1250) — compile-verified, matrix operand order checked by reading, not run on hardware
//
#include <hip/hip_runtime.h>
#include <hip/hip_bf16.h>
#include <cstddef>

typedef __attribute__((ext_vector_type(16))) _Float16 v16h;
typedef __attribute__((ext_vector_type(8)))  _Float16 v8h;
typedef __attribute__((ext_vector_type(8)))  float    v8f;

#define FEAT   128
#define HID    512
#define K1PAD  160           // 129 -> padded to multiple of 32
#define ROWS   16            // rows (batch) per workgroup == one WMMA M tile
#define BDIM   256           // 8 waves of 32
#define HSTR   520           // f16 LDS row stride (512+8): 16B aligned, bank-rotating
#define XSTR   168           // f16 LDS row stride for input layer (160+8)
#define FSTR   132           // f32 LDS row stride (128+4)

// ---------------------------------------------------------------------------
// Weight pre-pack: f32 row-major (Ksrc x N) -> f16 WMMA-native blocked layout.
// Block (kc, nt) = 32(K) x 16(N) tile, 512 halves = 1KB, block index nt*nK+kc.
// Within a block, lane L owns 16 contiguous halves at L*16:
//   lane<16 : col = nt*16+L,    halves 0..7 -> K kc*32+[0..7],  8..15 -> +[16..23]
//   lane>=16: col = nt*16+L-16, halves 0..7 -> K kc*32+[8..15], 8..15 -> +[24..31]
// ---------------------------------------------------------------------------
__global__ void pack_w_kernel(const float* __restrict__ W, _Float16* __restrict__ dst,
                              int Ksrc, int Kpad, int N)
{
    int total = Kpad * N;
    int tid = blockIdx.x * blockDim.x + threadIdx.x;
    if (tid >= total) return;
    int b    = tid >> 9;       // block of 512 halves
    int r    = tid & 511;
    int lane = r >> 4;
    int i    = r & 15;
    int nK   = Kpad >> 5;
    int nt   = b / nK;
    int kc   = b - nt * nK;
    int col  = nt * 16 + (lane & 15);
    int k;
    if (lane < 16) k = kc * 32 + ((i < 8) ? i       : 8 + i);
    else           k = kc * 32 + ((i < 8) ? 8 + i   : 16 + i);
    float v = (k < Ksrc && col < N) ? W[(size_t)k * N + col] : 0.0f;
    dst[tid] = (_Float16)v;
}

// ---------------------------------------------------------------------------
// Operand loaders
// ---------------------------------------------------------------------------
union H16 { v16h v; v8h h[2]; };

// A operand (16xK row-major f16 in LDS, row stride S halves), k-chunk kc.
// CDNA5 16-bit A layout: lane<16 row=lane holds K {0..7,16..23},
// lane>=16 row=lane-16 holds K {8..15,24..31}. Two 16B ds_load_b128.
static __device__ inline v16h lds_loadA(const _Float16* base, int S, int kc, int lane)
{
    int m  = lane & 15;
    int hi = lane >> 4;                 // 0 or 1
    const _Float16* p = base + m * S + kc * 32 + hi * 8;
    H16 u;
    u.h[0] = *(const v8h*)(p);          // K +0..7   (or +8..15)
    u.h[1] = *(const v8h*)(p + 16);     // K +16..23 (or +24..31)
    return u.v;
}

// B operand from pre-packed global blocks: lane's 16 halves are contiguous (32B).
static __device__ inline v16h glb_loadB(const _Float16* __restrict__ blocks, int blk, int lane)
{
    const _Float16* p = blocks + (size_t)blk * 512 + lane * 16;
    H16 u;
    u.h[0] = *(const v8h*)(p);
    u.h[1] = *(const v8h*)(p + 8);
    return u.v;
}

// Fast SiLU: x * rcp(1 + exp(-x)) -> v_mul, v_exp_f32, v_add, v_rcp_f32, v_mul.
// Avoids the IEEE-div lowering (div_scale/div_fmas Newton chain) on the
// critical inter-GEMM path.
static __device__ inline float silu(float x)
{
    return x * __builtin_amdgcn_rcpf(1.0f + __expf(-x));
}

// ---------------------------------------------------------------------------
// Persistent fused RK4 integrator. One 16-row block per workgroup, all steps.
// ---------------------------------------------------------------------------
__global__ __launch_bounds__(BDIM) void node_rk4_kernel(
    const float* __restrict__ x0,
    const float* __restrict__ b1, const float* __restrict__ b2, const float* __restrict__ b3,
    const _Float16* __restrict__ W1p, const _Float16* __restrict__ W2p, const _Float16* __restrict__ W3p,
    const int* __restrict__ nsp, float* __restrict__ out, int batch)
{
    __shared__ _Float16 sHA [ROWS * HSTR];   // layer-1 activations (f16)
    __shared__ _Float16 sHB [ROWS * HSTR];   // layer-2 activations (f16)
    __shared__ _Float16 sXin[ROWS * XSTR];   // [xc, t, 0-pad] input operand (f16)
    __shared__ float    sX  [ROWS * FSTR];   // ODE state (f32)
    __shared__ float    sK  [ROWS * FSTR];   // current stage derivative k (f32)
    __shared__ float    sAcc[ROWS * FSTR];   // RK4 weighted sum (f32)

    const int tid  = threadIdx.x;
    const int lane = tid & 31;
    const int wv   = tid >> 5;               // wave id 0..7
    const int row0 = blockIdx.x * ROWS;
    const int n_steps = *nsp;
    const size_t ostride = (size_t)batch * FEAT;

    // Epilogue coordinates (f32 C-matrix layout: lane<16 -> M 0..7, lane>=16 -> M 8..15)
    const int nlo   = lane & 15;
    const int mbase = (lane >> 4) * 8;

    // Load state block; emit trajectory frame 0.
    for (int i = tid; i < ROWS * FEAT; i += BDIM) {
        int m = i >> 7, n = i & 127;
        float v = x0[(size_t)(row0 + m) * FEAT + n];
        sX[m * FSTR + n] = v;
        out[(size_t)(row0 + m) * FEAT + n] = v;
    }
    __syncthreads();

    for (int step = 0; step < n_steps - 1; ++step) {
        const float t0   = (float)step       / (float)(n_steps - 1);
        const float t1   = (float)(step + 1) / (float)(n_steps - 1);
        const float dt   = t1 - t0;
        const float half = 0.5f * dt;

        for (int i = tid; i < ROWS * FSTR; i += BDIM) sAcc[i] = 0.0f;
        __syncthreads();

        for (int stage = 0; stage < 4; ++stage) {
            const float toff  = (stage == 0) ? 0.0f : (stage == 3) ? dt : half;
            const float cprev = toff;                                  // xc = x + cprev*k_prev
            const float wk    = (stage == 1 || stage == 2) ? dt / 3.0f : dt / 6.0f;
            const float ts    = t0 + toff;

            // Build f16 input operand: cols 0..127 = xc, col 128 = t, 129..159 = 0
            for (int i = tid; i < ROWS * K1PAD; i += BDIM) {
                int m = i / K1PAD, c = i - m * K1PAD;
                float v;
                if (c < FEAT) {
                    v = sX[m * FSTR + c];
                    if (stage != 0) v += cprev * sK[m * FSTR + c];
                } else {
                    v = (c == FEAT) ? ts : 0.0f;
                }
                sXin[m * XSTR + c] = (_Float16)v;
            }
            __syncthreads();

            // ---- GEMM1: (16x160) @ (160x512) -> SiLU -> sHA ----
            // k-outer, 4 rotating accumulators: A chunk loaded once per kc,
            // WMMA->WMMA D-reuse distance of 4 for pipelining.
            {
                v8f acc0 = {}, acc1 = {}, acc2 = {}, acc3 = {};
#pragma unroll
                for (int kc = 0; kc < 5; ++kc) {
                    v16h a  = lds_loadA(sXin, XSTR, kc, lane);
                    v16h bb0 = glb_loadB(W1p, (wv +  0) * 5 + kc, lane);
                    v16h bb1 = glb_loadB(W1p, (wv +  8) * 5 + kc, lane);
                    v16h bb2 = glb_loadB(W1p, (wv + 16) * 5 + kc, lane);
                    v16h bb3 = glb_loadB(W1p, (wv + 24) * 5 + kc, lane);
                    acc0 = __builtin_amdgcn_wmma_f32_16x16x32_f16(false, a, false, bb0, (short)0, acc0, false, false);
                    acc1 = __builtin_amdgcn_wmma_f32_16x16x32_f16(false, a, false, bb1, (short)0, acc1, false, false);
                    acc2 = __builtin_amdgcn_wmma_f32_16x16x32_f16(false, a, false, bb2, (short)0, acc2, false, false);
                    acc3 = __builtin_amdgcn_wmma_f32_16x16x32_f16(false, a, false, bb3, (short)0, acc3, false, false);
                }
                v8f accs[4] = {acc0, acc1, acc2, acc3};
#pragma unroll
                for (int t = 0; t < 4; ++t) {
                    const int n = (wv + t * 8) * 16 + nlo;
                    const float bias = b1[n];
#pragma unroll
                    for (int i = 0; i < 8; ++i)
                        sHA[(mbase + i) * HSTR + n] = (_Float16)silu(accs[t][i] + bias);
                }
            }
            __syncthreads();

            // ---- GEMM2: (16x512) @ (512x512) -> SiLU -> sHB ----
            {
                v8f acc0 = {}, acc1 = {}, acc2 = {}, acc3 = {};
                for (int kc = 0; kc < 16; ++kc) {
                    v16h a  = lds_loadA(sHA, HSTR, kc, lane);
                    v16h bb0 = glb_loadB(W2p, (wv +  0) * 16 + kc, lane);
                    v16h bb1 = glb_loadB(W2p, (wv +  8) * 16 + kc, lane);
                    v16h bb2 = glb_loadB(W2p, (wv + 16) * 16 + kc, lane);
                    v16h bb3 = glb_loadB(W2p, (wv + 24) * 16 + kc, lane);
                    acc0 = __builtin_amdgcn_wmma_f32_16x16x32_f16(false, a, false, bb0, (short)0, acc0, false, false);
                    acc1 = __builtin_amdgcn_wmma_f32_16x16x32_f16(false, a, false, bb1, (short)0, acc1, false, false);
                    acc2 = __builtin_amdgcn_wmma_f32_16x16x32_f16(false, a, false, bb2, (short)0, acc2, false, false);
                    acc3 = __builtin_amdgcn_wmma_f32_16x16x32_f16(false, a, false, bb3, (short)0, acc3, false, false);
                }
                v8f accs[4] = {acc0, acc1, acc2, acc3};
#pragma unroll
                for (int t = 0; t < 4; ++t) {
                    const int n = (wv + t * 8) * 16 + nlo;
                    const float bias = b2[n];
#pragma unroll
                    for (int i = 0; i < 8; ++i)
                        sHB[(mbase + i) * HSTR + n] = (_Float16)silu(accs[t][i] + bias);
                }
            }
            __syncthreads();

            // ---- GEMM3: (16x512) @ (512x128) -> k ; RK4 bookkeeping ----
            {
                v8f c = {};
                for (int kc = 0; kc < 16; ++kc) {
                    v16h a = lds_loadA(sHB, HSTR, kc, lane);
                    v16h b = glb_loadB(W3p, wv * 16 + kc, lane);
                    c = __builtin_amdgcn_wmma_f32_16x16x32_f16(false, a, false, b,
                                                               (short)0, c, false, false);
                }
                const int n = wv * 16 + nlo;       // 8 waves cover 128 cols
                const float bias = b3[n];
#pragma unroll
                for (int i = 0; i < 8; ++i) {
                    float kv = c[i] + bias;
                    int idx = (mbase + i) * FSTR + n;
                    sK[idx]    = kv;
                    sAcc[idx] += wk * kv;
                }
            }
            __syncthreads();
        } // stages

        // x += dt/6 * (k1 + 2k2 + 2k3 + k4) ; emit trajectory frame step+1
        float* dst = out + (size_t)(step + 1) * ostride;
        for (int i = tid; i < ROWS * FEAT; i += BDIM) {
            int m = i >> 7, n = i & 127;
            float xn = sX[m * FSTR + n] + sAcc[m * FSTR + n];
            sX[m * FSTR + n] = xn;
            dst[(size_t)(row0 + m) * FEAT + n] = xn;
        }
        __syncthreads();
    }
}

// ---------------------------------------------------------------------------
// Host launcher
// ---------------------------------------------------------------------------
extern "C" void kernel_launch(void* const* d_in, const int* in_sizes, int n_in,
                              void* d_out, int out_size, void* d_ws, size_t ws_size,
                              hipStream_t stream)
{
    const float* x  = (const float*)d_in[0];
    const float* W1 = (const float*)d_in[1];
    const float* b1 = (const float*)d_in[2];
    const float* W2 = (const float*)d_in[3];
    const float* b2 = (const float*)d_in[4];
    const float* W3 = (const float*)d_in[5];
    const float* b3 = (const float*)d_in[6];
    const int*   ns = (const int*)d_in[7];

    const int batch = in_sizes[0] / FEAT;        // 4096

    _Float16* W1p = (_Float16*)d_ws;             // 160*512 halves
    _Float16* W2p = W1p + K1PAD * HID;           // 512*512 halves
    _Float16* W3p = W2p + HID * HID;             // 512*128 halves
    // total workspace: (160*512 + 512*512 + 512*128)*2 = 819,200 bytes

    pack_w_kernel<<<(K1PAD * HID + 255) / 256, 256, 0, stream>>>(W1, W1p, FEAT + 1, K1PAD, HID);
    pack_w_kernel<<<(HID * HID + 255) / 256, 256, 0, stream>>>(W2, W2p, HID, HID, HID);
    pack_w_kernel<<<(HID * FEAT + 255) / 256, 256, 0, stream>>>(W3, W3p, HID, HID, FEAT);

    node_rk4_kernel<<<batch / ROWS, BDIM, 0, stream>>>(
        x, b1, b2, b3, W1p, W2p, W3p, ns, (float*)d_out, batch);
}
